// BatchHardLoss_26147760898823
// MI455X (gfx1250) — matrix operand order repaired and checked by reading, not compiled, not run on hardware
//
#include <hip/hip_runtime.h>
#include <hip/hip_bf16.h>

// Problem constants (match reference)
#define B_N   8192
#define D_K   256
#define GAMMA 0.001f

typedef __bf16 bf16x8  __attribute__((ext_vector_type(8)));
typedef __bf16 bf16x16 __attribute__((ext_vector_type(16)));
typedef float  f32x8   __attribute__((ext_vector_type(8)));

// ---------------------------------------------------------------------------
// Kernel 0: convert f32 inputs -> bf16 (workspace) and zero the row-sum arrays
// ---------------------------------------------------------------------------
__global__ __launch_bounds__(256) void prep_kernel(const float* __restrict__ x,
                                                   __bf16* __restrict__ xb,
                                                   float* __restrict__ psum,
                                                   float* __restrict__ nsum) {
    int i = blockIdx.x * 256 + threadIdx.x;
    if (i < B_N * D_K) xb[i] = (__bf16)x[i];
    if (i < B_N) { psum[i] = 0.0f; nsum[i] = 0.0f; }
}

// ---------------------------------------------------------------------------
// Kernel 1: fused Gram-tile + masked-exp epilogue.
// Block = 256 threads = 8 waves; block tile = 128 rows x 128 cols.
// Wave grid: 4 along M (32 rows each) x 2 along N (64 cols each).
// Each wave: 2x4 accumulators of v_wmma_f32_16x16x32_bf16 over K=256.
// ---------------------------------------------------------------------------
__global__ __launch_bounds__(256) void gram_loss_tile(const __bf16* __restrict__ X,
                                                      const int*    __restrict__ tgt,
                                                      float* __restrict__ psum,
                                                      float* __restrict__ nsum) {
    const int lane = threadIdx.x & 31;
    const int wave = threadIdx.x >> 5;
    const int wm   = wave & 3;          // wave position along M (0..3)
    const int wn   = wave >> 2;         // wave position along N (0..1)
    const int rbase = blockIdx.y * 128 + wm * 32;
    const int cbase = blockIdx.x * 128 + wn * 64;
    const int l16 = lane & 15;
    const int h   = lane >> 4;          // which 16-lane half of the wave

    f32x8 acc[2][4] = {};

    for (int kc = 0; kc < D_K; kc += 32) {
        // A fragments (16x32 bf16, ISA A-layout): lane holds row rbase+mi*16+l16,
        // K chunks [kc+8h .. +7] and [kc+16+8h .. +7]  (two 16B loads)
        bf16x16 Af[2];
#pragma unroll
        for (int mi = 0; mi < 2; ++mi) {
            const __bf16* p = X + (size_t)(rbase + mi * 16 + l16) * D_K + kc + h * 8;
            bf16x8 lo = *(const bf16x8*)p;
            bf16x8 hi = *(const bf16x8*)(p + 16);
            Af[mi] = __builtin_shufflevector(lo, hi, 0,1,2,3,4,5,6,7,8,9,10,11,12,13,14,15);
        }
        // B fragments (32x16 bf16, column-major == rows of X for Gram):
        // lane holds "column" row cbase+ni*16+l16, K chunks [kc+8h .. +7] and [kc+16+8h .. +7]
        bf16x16 Bf[4];
#pragma unroll
        for (int ni = 0; ni < 4; ++ni) {
            const __bf16* p = X + (size_t)(cbase + ni * 16 + l16) * D_K + kc + h * 8;
            bf16x8 lo = *(const bf16x8*)p;
            bf16x8 hi = *(const bf16x8*)(p + 16);
            Bf[ni] = __builtin_shufflevector(lo, hi, 0,1,2,3,4,5,6,7,8,9,10,11,12,13,14,15);
        }
#pragma unroll
        for (int mi = 0; mi < 2; ++mi)
#pragma unroll
            for (int ni = 0; ni < 4; ++ni)
                acc[mi][ni] = __builtin_amdgcn_wmma_f32_16x16x32_bf16(
                    /*neg_a=*/false, Af[mi], /*neg_b=*/false, Bf[ni],
                    /*c_mod=*/(short)0, acc[mi][ni],
                    /*reuse_a=*/false, /*reuse_b=*/false);
    }

    // Fused epilogue. C layout: lane L, VGPR v -> row = base + 8*(L/16) + v,
    // col = colbase + (L&15). Build per-row masked exp sums, reduce the 16
    // columns held across each 16-lane half, then one lane atomically adds.
#pragma unroll
    for (int mi = 0; mi < 2; ++mi) {
        const int rows0 = rbase + mi * 16 + h * 8;   // 8 consecutive rows in this half
        int trow[8];
#pragma unroll
        for (int v = 0; v < 8; ++v) trow[v] = tgt[rows0 + v];

        float prow[8] = {0,0,0,0,0,0,0,0};
        float nrow[8] = {0,0,0,0,0,0,0,0};
#pragma unroll
        for (int ni = 0; ni < 4; ++ni) {
            const int c  = cbase + ni * 16 + l16;
            const int tc = tgt[c];
#pragma unroll
            for (int v = 0; v < 8; ++v) {
                float w = acc[mi][ni][v] * GAMMA;
                w = fminf(16.0f, fmaxf(-16.0f, w));
                const int  r    = rows0 + v;
                const bool same = (trow[v] == tc);
                const bool self = (r == c);
                prow[v] += (same && !self) ? __expf(-w) : 0.0f;
                nrow[v] += (!same)         ? __expf( w) : 0.0f;
            }
        }
#pragma unroll
        for (int v = 0; v < 8; ++v) {
            float p = prow[v], n = nrow[v];
            // butterfly within each 16-lane half (xor masks 1,2,4,8 never cross bit 4)
            for (int m = 1; m < 16; m <<= 1) {
                p += __shfl_xor(p, m, 32);
                n += __shfl_xor(n, m, 32);
            }
            if (l16 == 0) {
                atomicAdd(&psum[rows0 + v], p);
                atomicAdd(&nsum[rows0 + v], n);
            }
        }
    }
}

// ---------------------------------------------------------------------------
// Kernel 2: per_row = log(pos*neg); output = mean(per_row). Single block.
// ---------------------------------------------------------------------------
__global__ __launch_bounds__(256) void finalize_kernel(const float* __restrict__ psum,
                                                       const float* __restrict__ nsum,
                                                       float* __restrict__ out) {
    __shared__ float s[256];
    float acc = 0.0f;
    for (int r = threadIdx.x; r < B_N; r += 256)
        acc += logf(psum[r] * nsum[r]);
    s[threadIdx.x] = acc;
    __syncthreads();
    for (int off = 128; off > 0; off >>= 1) {
        if (threadIdx.x < off) s[threadIdx.x] += s[threadIdx.x + off];
        __syncthreads();
    }
    if (threadIdx.x == 0) out[0] = s[0] / (float)B_N;
}

// ---------------------------------------------------------------------------
extern "C" void kernel_launch(void* const* d_in, const int* in_sizes, int n_in,
                              void* d_out, int out_size, void* d_ws, size_t ws_size,
                              hipStream_t stream) {
    const float* x   = (const float*)d_in[0];   // [8192, 256] f32
    const int*   tgt = (const int*)d_in[1];     // [8192] i32
    float*       out = (float*)d_out;           // scalar f32

    // Workspace layout: bf16 X (4 MB) | psum (32 KB) | nsum (32 KB)
    __bf16* xb   = (__bf16*)d_ws;
    float*  psum = (float*)((char*)d_ws + (size_t)B_N * D_K * sizeof(__bf16));
    float*  nsum = psum + B_N;

    prep_kernel<<<(B_N * D_K) / 256, 256, 0, stream>>>(x, xb, psum, nsum);

    dim3 grid(B_N / 128, B_N / 128);  // 64 x 64 tiles of 128x128
    gram_loss_tile<<<grid, 256, 0, stream>>>(xb, tgt, psum, nsum);

    finalize_kernel<<<1, 256, 0, stream>>>(psum, nsum, out);
}
